// AttentionBlock_627065226000
// MI455X (gfx1250) — matrix operand order repaired and checked by reading, not compiled, not run on hardware
//
#include <hip/hip_runtime.h>

typedef __attribute__((ext_vector_type(16))) __bf16 v16bf;
typedef __attribute__((ext_vector_type(8)))  float  v8f;

#define CB 256   // channels
#define DB 32    // head dim (C/8)
#define NB 4096  // tokens (H*W)
#define BB 4     // batch

#if defined(__HIP_DEVICE_COMPILE__) && \
    __has_builtin(__builtin_amdgcn_global_load_async_to_lds_b128) && \
    __has_builtin(__builtin_amdgcn_s_wait_asynccnt)
#define HAVE_ASYNC_LDS 1
#else
#define HAVE_ASYNC_LDS 0
#endif

__device__ __forceinline__ unsigned pack_bf16(float a, float b) {
  unsigned ua = __builtin_bit_cast(unsigned, a);
  unsigned ub = __builtin_bit_cast(unsigned, b);
  ua = (ua + 0x7FFFu + ((ua >> 16) & 1u)) >> 16;   // RNE f32 -> bf16
  ub = (ub + 0x7FFFu + ((ub >> 16) & 1u)) >> 16;
  return ua | (ub << 16);
}

__device__ __forceinline__ unsigned short bf16_1(float a) {
  unsigned ua = __builtin_bit_cast(unsigned, a);
  return (unsigned short)((ua + 0x7FFFu + ((ua >> 16) & 1u)) >> 16);
}

#if HAVE_ASYNC_LDS
typedef int v4i_vs __attribute__((vector_size(16)));   // GCC-style V4i (per diag)
__device__ __forceinline__ void async_copy_b128(const void* gsrc, void* ldst) {
  __builtin_amdgcn_global_load_async_to_lds_b128(
      (__attribute__((address_space(1))) v4i_vs*)gsrc,
      (__attribute__((address_space(3))) v4i_vs*)ldst, 0, 0);
}
#endif

// ---------------------------------------------------------------------------
// Kernel 0a: x (f32 [b][c][n]) -> bf16 transposed x^T [b][n][c]
// (c-pairs adjacent => WMMA B-operand dwords for all three projections)
// ---------------------------------------------------------------------------
__global__ __launch_bounds__(256) void x_to_bf16t_kernel(
    const float* __restrict__ x, unsigned short* __restrict__ xt) {
  __shared__ unsigned short tile[64][66];          // pad to dodge bank conflicts
  const int b = blockIdx.z, c0 = blockIdx.y * 64, n0 = blockIdx.x * 64;
  const float* xp = x + ((size_t)b * CB + c0) * NB + n0;
#pragma unroll
  for (int rr = 0; rr < 16; ++rr) {
    int idx = rr * 256 + threadIdx.x;
    int cl = idx >> 6, nl = idx & 63;
    tile[cl][nl] = bf16_1(xp[(size_t)cl * NB + nl]);   // coalesced in n
  }
  __syncthreads();
  unsigned short* op = xt + ((size_t)b * NB + n0) * CB + c0;
#pragma unroll
  for (int rr = 0; rr < 16; ++rr) {
    int idx = rr * 256 + threadIdx.x;
    int nl = idx >> 6, cl = idx & 63;
    op[(size_t)nl * CB + cl] = tile[cl][nl];           // coalesced in c
  }
}

// ---------------------------------------------------------------------------
// Kernel 0b: weights f32 -> bf16 row-major (A-operand packing: c-pairs adjacent)
// ---------------------------------------------------------------------------
__global__ __launch_bounds__(256) void w_to_bf16_kernel(
    const float* __restrict__ qw, const float* __restrict__ kw,
    const float* __restrict__ vw,
    unsigned short* __restrict__ qwb, unsigned short* __restrict__ kwb,
    unsigned short* __restrict__ vwb) {
  int t = blockIdx.x * 256 + threadIdx.x;          // 0..81919
  if (t < 8192)        qwb[t]         = bf16_1(qw[t]);
  else if (t < 16384)  kwb[t - 8192]  = bf16_1(kw[t - 8192]);
  else                 vwb[t - 16384] = bf16_1(vw[t - 16384]);
}

// ---------------------------------------------------------------------------
// Kernel 1: q/k projection GEMM via WMMA.
// Wave = 16 tokens x (32 q-dims + 32 k-dims). D = W(16d x 32c) * X^T(32c x 16n).
// ---------------------------------------------------------------------------
__global__ __launch_bounds__(128) void proj_qk_wmma_kernel(
    const unsigned short* __restrict__ xt,   // [b][n][c] bf16
    const unsigned* __restrict__ qwp,        // [32][128] u32 (bf16 pairs)
    const unsigned* __restrict__ kwp,
    const float* __restrict__ qb, const float* __restrict__ kb,
    unsigned* __restrict__ qo, unsigned* __restrict__ ko) {
  const int wave = threadIdx.x >> 5, lane = threadIdx.x & 31;
  const int ln = lane & 15, hi = lane >> 4;
  const int gw = blockIdx.x * 4 + wave;            // 0..1023
  const int b = gw >> 8;
  const int n0 = (gw & 255) * 16;

  union Bf16x16 { v16bf v; unsigned u[8]; };
  const v8f vz = (v8f){0.f,0.f,0.f,0.f,0.f,0.f,0.f,0.f};
  v8f aq0 = vz, aq1 = vz, ak0 = vz, ak1 = vz;

  const unsigned* xrow = (const unsigned*)xt + ((size_t)b * NB + n0 + ln) * (CB / 2);

  for (int ks = 0; ks < 8; ++ks) {                 // K = 32 channels per step
    Bf16x16 xb;                                    // B operand: n=token, K=c
#pragma unroll
    for (int r = 0; r < 8; ++r) xb.u[r] = xrow[ks * 16 + ((r < 4) ? (hi * 4 + r) : (8 + hi * 4 + (r - 4)))];
    Bf16x16 a;
#pragma unroll
    for (int r = 0; r < 8; ++r) {                  // A operand pair index
      int kp = (r < 4) ? (hi * 4 + r) : (8 + hi * 4 + (r - 4));
      a.u[r] = qwp[(size_t)ln * 128 + ks * 16 + kp];
    }
    aq0 = __builtin_amdgcn_wmma_f32_16x16x32_bf16(false, a.v, false, xb.v, (short)0, aq0, false, false);
#pragma unroll
    for (int r = 0; r < 8; ++r) {
      int kp = (r < 4) ? (hi * 4 + r) : (8 + hi * 4 + (r - 4));
      a.u[r] = qwp[(size_t)(16 + ln) * 128 + ks * 16 + kp];
    }
    aq1 = __builtin_amdgcn_wmma_f32_16x16x32_bf16(false, a.v, false, xb.v, (short)0, aq1, false, false);
#pragma unroll
    for (int r = 0; r < 8; ++r) {
      int kp = (r < 4) ? (hi * 4 + r) : (8 + hi * 4 + (r - 4));
      a.u[r] = kwp[(size_t)ln * 128 + ks * 16 + kp];
    }
    ak0 = __builtin_amdgcn_wmma_f32_16x16x32_bf16(false, a.v, false, xb.v, (short)0, ak0, false, false);
#pragma unroll
    for (int r = 0; r < 8; ++r) {
      int kp = (r < 4) ? (hi * 4 + r) : (8 + hi * 4 + (r - 4));
      a.u[r] = kwp[(size_t)(16 + ln) * 128 + ks * 16 + kp];
    }
    ak1 = __builtin_amdgcn_wmma_f32_16x16x32_bf16(false, a.v, false, xb.v, (short)0, ak1, false, false);
  }

  // D rows = d = ct*16 + r + 8*hi, cols = token n0+ln. Add bias, pack bf16 pairs.
  const int n = n0 + ln;
  unsigned* qrow = qo + ((size_t)b * NB + n) * 16;
  unsigned* krow = ko + ((size_t)b * NB + n) * 16;
#pragma unroll
  for (int rp = 0; rp < 4; ++rp) {
    int d0 = 2 * rp + 8 * hi;
    qrow[4 * hi + rp]     = pack_bf16(aq0[2 * rp] + qb[d0],      aq0[2 * rp + 1] + qb[d0 + 1]);
    qrow[8 + 4 * hi + rp] = pack_bf16(aq1[2 * rp] + qb[16 + d0], aq1[2 * rp + 1] + qb[17 + d0]);
    krow[4 * hi + rp]     = pack_bf16(ak0[2 * rp] + kb[d0],      ak0[2 * rp + 1] + kb[d0 + 1]);
    krow[8 + 4 * hi + rp] = pack_bf16(ak1[2 * rp] + kb[16 + d0], ak1[2 * rp + 1] + kb[17 + d0]);
  }
}

// ---------------------------------------------------------------------------
// Kernel 2: v projection GEMM via WMMA. Wave = 16 tokens x 256 out-channels.
// ---------------------------------------------------------------------------
__global__ __launch_bounds__(128) void proj_v_wmma_kernel(
    const unsigned short* __restrict__ xt,   // [b][n][c] bf16
    const unsigned* __restrict__ vwp,        // [256][128] u32 (bf16 pairs)
    const float* __restrict__ vb,
    unsigned short* __restrict__ vo) {       // [b][c][n] bf16
  const int wave = threadIdx.x >> 5, lane = threadIdx.x & 31;
  const int ln = lane & 15, hi = lane >> 4;
  const int gw = blockIdx.x * 4 + wave;
  const int b = gw >> 8;
  const int n0 = (gw & 255) * 16;

  union Bf16x16 { v16bf v; unsigned u[8]; };
  v8f acc[16];
#pragma unroll
  for (int t = 0; t < 16; ++t) acc[t] = (v8f){0.f,0.f,0.f,0.f,0.f,0.f,0.f,0.f};

  const unsigned* xrow = (const unsigned*)xt + ((size_t)b * NB + n0 + ln) * (CB / 2);

  for (int ks = 0; ks < 8; ++ks) {
    Bf16x16 xb;                                    // B operand reused by 16 tiles
#pragma unroll
    for (int r = 0; r < 8; ++r) xb.u[r] = xrow[ks * 16 + ((r < 4) ? (hi * 4 + r) : (8 + hi * 4 + (r - 4)))];
#pragma unroll
    for (int ct = 0; ct < 16; ++ct) {
      Bf16x16 a;
#pragma unroll
      for (int r = 0; r < 8; ++r) {
        int kp = (r < 4) ? (hi * 4 + r) : (8 + hi * 4 + (r - 4));
        a.u[r] = vwp[(size_t)(ct * 16 + ln) * 128 + ks * 16 + kp];
      }
      acc[ct] = __builtin_amdgcn_wmma_f32_16x16x32_bf16(false, a.v, false, xb.v,
                                                        (short)0, acc[ct], false, false);
    }
  }

  const int n = n0 + ln;
#pragma unroll
  for (int ct = 0; ct < 16; ++ct) {
#pragma unroll
    for (int r = 0; r < 8; ++r) {
      int c = ct * 16 + r + 8 * hi;
      vo[((size_t)b * CB + c) * NB + n] = bf16_1(acc[ct][r] + vb[c]);
    }
  }
}

// ---------------------------------------------------------------------------
// Kernel 3: flash attention. 4 waves/block; wave = 16 queries x 256 channels.
// Double-buffered K/V tiles staged with async-to-LDS (ASYNCcnt) when available.
// Per 32-key tile: 2 WMMA (scores) + online softmax + 16 WMMA (O^T += V*P^T).
// ---------------------------------------------------------------------------
__global__ __launch_bounds__(128) void flash_attn_kernel(
    const float* __restrict__ x,
    const unsigned* __restrict__ qws,        // [B][N][16] u32 (bf16 pairs)
    const unsigned* __restrict__ kws,        // [B][N][16] u32
    const unsigned short* __restrict__ vws,  // [B][C][N] bf16
    const float* __restrict__ gamma,
    float* __restrict__ out) {
  __shared__ __align__(16) unsigned short kbuf[2][32][32];   // [par][j_local][d]
  __shared__ __align__(16) unsigned short vbuf[2][CB][32];   // [par][c][j_local]
  __shared__ float    sbuf[4][16][32];                       // per-wave scores [i][j]
  __shared__ unsigned pbuf[4][16][16];                       // per-wave P bf16x2

  const int wave = threadIdx.x >> 5;
  const int lane = threadIdx.x & 31;
  const int ln = lane & 15;
  const int hi = lane >> 4;
  const int b = blockIdx.x >> 6;
  const int qbase = (blockIdx.x & 63) * 64 + wave * 16;

  union Bf16x16 { v16bf v; unsigned u[8]; };

  // Q tile as WMMA A operand (16 queries x 32 d), loaded once, reused 128x.
  Bf16x16 qA;
  {
    const unsigned* qrow = qws + ((size_t)b * NB + qbase + ln) * 16;
#pragma unroll
    for (int r = 0; r < 8; ++r) {
      int kp = hi * 8 + r;
      qA.u[r] = qrow[kp];
    }
  }

  v8f acc[16];                                   // O^T accumulators: 16 c-tiles
#pragma unroll
  for (int t = 0; t < 16; ++t)
    acc[t] = (v8f){0.f,0.f,0.f,0.f,0.f,0.f,0.f,0.f};

  float row_m = -1.0e30f;                        // stats for query row qbase+ln
  float row_l = 0.0f;

#if HAVE_ASYNC_LDS
  auto issue_tile = [&](int j0, int par) {       // 9 async b128 per thread/wave
    const char* ks = (const char*)(kws + ((size_t)b * NB + j0) * 16) + threadIdx.x * 16;
    async_copy_b128(ks, (char*)(&kbuf[par][0][0]) + threadIdx.x * 16);
#pragma unroll
    for (int rr = 0; rr < 2; ++rr) {
      int c = threadIdx.x * 2 + rr;
      const unsigned short* vs = vws + ((size_t)b * CB + c) * NB + j0;
#pragma unroll
      for (int ch = 0; ch < 4; ++ch)
        async_copy_b128(vs + ch * 8, &vbuf[par][c][ch * 8]);
    }
  };
  issue_tile(0, 0);
#endif

  const int NT = NB / 32;
  for (int jt = 0; jt < NT; ++jt) {
    const int par = jt & 1;
    const int j0 = jt * 32;
#if HAVE_ASYNC_LDS
    if (jt + 1 < NT) {
      issue_tile(j0 + 32, par ^ 1);              // prefetch next tile
      __builtin_amdgcn_s_wait_asynccnt(9);       // current tile done (in-order)
    } else {
      __builtin_amdgcn_s_wait_asynccnt(0);
    }
#else
    {
      const unsigned* ksrc = kws + ((size_t)b * NB + j0) * 16;
      unsigned* kdst = (unsigned*)&kbuf[par][0][0];
#pragma unroll
      for (int r = 0; r < 4; ++r) {
        int idx = r * 128 + threadIdx.x;
        kdst[idx] = ksrc[idx];
      }
#pragma unroll
      for (int rr = 0; rr < 2; ++rr) {
        int c = threadIdx.x * 2 + rr;
        const uint4* s = (const uint4*)(vws + ((size_t)b * CB + c) * NB + j0);
        uint4* d4 = (uint4*)&vbuf[par][c][0];
        d4[0] = s[0]; d4[1] = s[1]; d4[2] = s[2]; d4[3] = s[3];
      }
      if (j0 + 32 < NB)
        __builtin_prefetch(vws + ((size_t)b * CB + threadIdx.x * 2) * NB + j0 + 32, 0, 1);
    }
#endif
    __syncthreads();

    // ---- scores: S(16q x 32j) = Q(16x32d) * K^T via 2 WMMAs ----
    v8f s0 = (v8f){0.f,0.f,0.f,0.f,0.f,0.f,0.f,0.f};
    v8f s1 = (v8f){0.f,0.f,0.f,0.f,0.f,0.f,0.f,0.f};
    {
      Bf16x16 k0, k1;                            // B operand: n=j, K=d
#pragma unroll
      for (int r = 0; r < 8; ++r) {
        k0.u[r] = *(const unsigned*)&kbuf[par][ln][hi * 16 + 2 * r];
        k1.u[r] = *(const unsigned*)&kbuf[par][16 + ln][hi * 16 + 2 * r];
      }
      s0 = __builtin_amdgcn_wmma_f32_16x16x32_bf16(false, qA.v, false, k0.v,
                                                   (short)0, s0, false, false);
      s1 = __builtin_amdgcn_wmma_f32_16x16x32_bf16(false, qA.v, false, k1.v,
                                                   (short)0, s1, false, false);
    }
    // spill C-layout tiles to LDS row-major [i][j] for the softmax transpose
#pragma unroll
    for (int r = 0; r < 8; ++r) {
      sbuf[wave][r + 8 * hi][ln]      = s0[r];
      sbuf[wave][r + 8 * hi][16 + ln] = s1[r];
    }
    __threadfence_block();                       // same-wave LDS RAW ordering

    // ---- online softmax: lanes l and l+16 co-own query row ln ----
    float sv[16];
#pragma unroll
    for (int jj = 0; jj < 16; ++jj) sv[jj] = sbuf[wave][ln][hi * 16 + jj];
    float pmax = sv[0];
#pragma unroll
    for (int jj = 1; jj < 16; ++jj) pmax = fmaxf(pmax, sv[jj]);
    pmax = fmaxf(pmax, __shfl_xor(pmax, 16, 32));
    float new_m = fmaxf(row_m, pmax);
    float scale = __expf(row_m - new_m);
    float psum = 0.0f;
#pragma unroll
    for (int jj = 0; jj < 8; ++jj) {
      float e0 = __expf(sv[2 * jj]     - new_m);
      float e1 = __expf(sv[2 * jj + 1] - new_m);
      psum += e0 + e1;
      pbuf[wave][ln][hi * 8 + jj] = pack_bf16(e0, e1);   // P row ln, bf16
    }
    psum += __shfl_xor(psum, 16, 32);
    row_l = row_l * scale + psum;
    row_m = new_m;
    __threadfence_block();

    // acc columns are query i = ln -> per-lane scalar rescale
#pragma unroll
    for (int t = 0; t < 16; ++t) acc[t] = acc[t] * scale;

    // ---- O^T(16c x 16i) += V(16c x 32j) * P^T(32j x 16i), 16 c-tiles ----
    Bf16x16 pop;                                 // B operand: n=i, K=j
#pragma unroll
    for (int r = 0; r < 8; ++r) pop.u[r] = pbuf[wave][ln][hi * 8 + r];
#pragma unroll
    for (int ct = 0; ct < 16; ++ct) {
      Bf16x16 vop;                               // A operand: m=c_local, K=j
#pragma unroll
      for (int r = 0; r < 8; ++r) {
        int k2 = hi * 16 + 2 * r;
        vop.u[r] = *(const unsigned*)&vbuf[par][ct * 16 + ln][k2];
      }
      acc[ct] = __builtin_amdgcn_wmma_f32_16x16x32_bf16(false, vop.v, false, pop.v,
                                                        (short)0, acc[ct], false, false);
    }
    __syncthreads();                             // all waves done with buf[par]
  }

  // ---- finalize: out = x + gamma * O / l  (stores coalesced in n) ----
  float inv_l = 1.0f / row_l;                    // column i = ln for this lane
  float g = gamma[0];
  int n = qbase + ln;
#pragma unroll
  for (int ct = 0; ct < 16; ++ct) {
#pragma unroll
    for (int r = 0; r < 8; ++r) {
      int c = ct * 16 + r + 8 * hi;
      size_t off = ((size_t)b * CB + c) * NB + n;
      out[off] = fmaf(g * inv_l, acc[ct][r], x[off]);
    }
  }
}

// ---------------------------------------------------------------------------
extern "C" void kernel_launch(void* const* d_in, const int* in_sizes, int n_in,
                              void* d_out, int out_size, void* d_ws, size_t ws_size,
                              hipStream_t stream) {
  (void)in_sizes; (void)n_in; (void)out_size; (void)ws_size;
  const float* x     = (const float*)d_in[0];
  const float* qw    = (const float*)d_in[1];
  const float* qb    = (const float*)d_in[2];
  const float* kw    = (const float*)d_in[3];
  const float* kb    = (const float*)d_in[4];
  const float* vw    = (const float*)d_in[5];
  const float* vb    = (const float*)d_in[6];
  const float* gamma = (const float*)d_in[7];
  float* out = (float*)d_out;

  // workspace partition (~18.2 MB):
  // q 1MB | k 1MB | v 8MB | x^T 8MB | qw 16KB | kw 16KB | vw 128KB  (all bf16)
  char* ws = (char*)d_ws;
  unsigned*       q_ws  = (unsigned*)(ws);
  unsigned*       k_ws  = (unsigned*)(ws + ((size_t)1 << 20));
  unsigned short* v_ws  = (unsigned short*)(ws + ((size_t)2 << 20));
  unsigned short* xt_ws = (unsigned short*)(ws + ((size_t)10 << 20));
  unsigned*       qw_pk = (unsigned*)(ws + ((size_t)18 << 20));
  unsigned*       kw_pk = (unsigned*)(ws + ((size_t)18 << 20) + (1 << 14));
  unsigned*       vw_pk = (unsigned*)(ws + ((size_t)18 << 20) + (2 << 14));

  x_to_bf16t_kernel<<<dim3(64, 4, 4), 256, 0, stream>>>(x, xt_ws);
  w_to_bf16_kernel<<<320, 256, 0, stream>>>(qw, kw, vw,
      (unsigned short*)qw_pk, (unsigned short*)kw_pk, (unsigned short*)vw_pk);
  proj_qk_wmma_kernel<<<256, 128, 0, stream>>>(xt_ws, qw_pk, kw_pk, qb, kb, q_ws, k_ws);
  proj_v_wmma_kernel<<<256, 128, 0, stream>>>(xt_ws, vw_pk, vb, v_ws);
  flash_attn_kernel<<<BB * (NB / 64), 128, 0, stream>>>(x, q_ws, k_ws, v_ws, gamma, out);
}
